// SparseMoE_65833258713206
// MI455X (gfx1250) — compile-verified, matrix operand order checked by reading, not run on hardware
//
#include <hip/hip_runtime.h>
#include <math.h>

// Problem constants (match reference).
#define Bdim 2
#define Sdim 2048
#define Ddim 1024
#define Hdim 2048
#define Edim 8
#define Ntok (Bdim * Sdim)   // 4096 tokens

typedef __attribute__((ext_vector_type(16))) __bf16 v16bf;
typedef __attribute__((ext_vector_type(8)))  __bf16 v8bf;
typedef __attribute__((ext_vector_type(8)))  float  v8f;

__device__ __forceinline__ __bf16 f32_to_bf16_rne(float f) {
  unsigned u = __builtin_bit_cast(unsigned, f);
  unsigned r = u + 0x7FFFu + ((u >> 16) & 1u);
  unsigned short hs = (unsigned short)(r >> 16);
  return __builtin_bit_cast(__bf16, hs);
}

// ---------------------------------------------------------------------------
// 1) init: zero expert counters and the output (atomic-accumulated later).
// ---------------------------------------------------------------------------
__global__ void moe_init_kernel(int* __restrict__ cnt, float* __restrict__ out, size_t n) {
  size_t i = (size_t)blockIdx.x * blockDim.x + threadIdx.x;
  if (i < Edim) cnt[i] = 0;
  if (i < n) out[i] = 0.0f;
}

// ---------------------------------------------------------------------------
// 2) x: fp32 -> bf16 (row-major [Ntok, D], A-operand side).
// ---------------------------------------------------------------------------
__global__ void moe_cvt_x_kernel(const float* __restrict__ x, __bf16* __restrict__ xb, size_t n) {
  size_t i = (size_t)blockIdx.x * blockDim.x + threadIdx.x;
  if (i < n) xb[i] = f32_to_bf16_rne(x[i]);
}

// ---------------------------------------------------------------------------
// 3) Pack a [E, Kd, Nd] fp32 weight into bf16 WMMA-B tiles.
// Per (e, n_tile, k_chunk): 512 bf16 laid out so lane L's 16 operands are
// contiguous. Lane L covers column n = n_tile*16 + (L&15); kb = (L<16)?0:8;
// element j -> K = k_chunk*32 + kb + (j<8 ? j : 16+(j-8)).
// This mirrors the CDNA5 16-bit B-operand VGPR striping, so the GEMM hot loop
// reads each B tile with two b128 loads per lane.
// ---------------------------------------------------------------------------
__global__ void moe_pack_w_kernel(const float* __restrict__ w, __bf16* __restrict__ wp,
                                  int Kd, int Nd) {
  size_t idx = (size_t)blockIdx.x * blockDim.x + threadIdx.x;
  size_t total = (size_t)Edim * Kd * Nd;
  if (idx >= total) return;
  int j    = (int)(idx & 15);
  int lane = (int)((idx >> 4) & 31);
  size_t tile = idx >> 9;
  int KC = Kd >> 5, NT = Nd >> 4;
  int kc = (int)(tile % KC);
  size_t rest = tile / KC;
  int nt = (int)(rest % NT);
  int e  = (int)(rest / NT);
  int n  = nt * 16 + (lane & 15);
  int kb = (lane < 16) ? 0 : 8;
  int k  = kc * 32 + kb + ((j < 8) ? j : (16 + (j - 8)));
  wp[idx] = f32_to_bf16_rne(w[((size_t)e * Kd + k) * Nd + n]);
}

// ---------------------------------------------------------------------------
// 4) Gating: logits, top-2, softmax over the 2, append to per-expert lists.
// list entry p of expert e: token id, h-slot (= 2*t + which_of_top2), weight.
// ---------------------------------------------------------------------------
__global__ void moe_gate_kernel(const float* __restrict__ x, const float* __restrict__ gw,
                                int* __restrict__ cnt, int* __restrict__ ltok,
                                int* __restrict__ lslot, float* __restrict__ lw) {
  int t = blockIdx.x * blockDim.x + threadIdx.x;
  if (t >= Ntok) return;
  float acc[Edim];
#pragma unroll
  for (int e = 0; e < Edim; ++e) acc[e] = 0.0f;
  const float* xr = x + (size_t)t * Ddim;
  for (int d = 0; d < Ddim; ++d) {
    float xv = xr[d];
#pragma unroll
    for (int e = 0; e < Edim; ++e) acc[e] = fmaf(xv, gw[d * Edim + e], acc[e]);
  }
  int e0 = 0; float v0 = acc[0];
#pragma unroll
  for (int e = 1; e < Edim; ++e) { if (acc[e] > v0) { v0 = acc[e]; e0 = e; } }
  int e1 = -1; float v1 = -3.4e38f;
#pragma unroll
  for (int e = 0; e < Edim; ++e) { if (e != e0 && acc[e] > v1) { v1 = acc[e]; e1 = e; } }
  // softmax over {v0, v1} with v0 as max
  float g1 = __expf(v1 - v0);
  float inv = 1.0f / (1.0f + g1);
  float w0 = inv, w1v = g1 * inv;
  int p0 = atomicAdd(&cnt[e0], 1);
  ltok[e0 * Ntok + p0] = t; lslot[e0 * Ntok + p0] = 2 * t;     lw[e0 * Ntok + p0] = w0;
  int p1 = atomicAdd(&cnt[e1], 1);
  ltok[e1 * Ntok + p1] = t; lslot[e1 * Ntok + p1] = 2 * t + 1; lw[e1 * Ntok + p1] = w1v;
}

// ---------------------------------------------------------------------------
// 5) GEMM1: h = silu(x@w1 + b1) * (x@w2 + b2) for each routed (token, expert).
// Block: 256 threads = 8 wave32; wave owns one 16x16 N-subtile for BOTH the
// w1 and w2 paths (two accumulators). Grid: (Ntok/16 tiles, H/128, E).
// ---------------------------------------------------------------------------
__global__ void __launch_bounds__(256) moe_gemm1_kernel(
    const __bf16* __restrict__ x_bf, const __bf16* __restrict__ w1p,
    const __bf16* __restrict__ w2p, const float* __restrict__ b1,
    const float* __restrict__ b2, const int* __restrict__ cnt,
    const int* __restrict__ ltok, const int* __restrict__ lslot,
    __bf16* __restrict__ h_bf) {
  const int e = blockIdx.z;
  const int count = cnt[e];
  const int m0 = blockIdx.x * 16;
  if (m0 >= count) return;                 // block-uniform early exit
  const int lane = threadIdx.x & 31;
  const int wave = threadIdx.x >> 5;
  const int ntile = blockIdx.y * 8 + wave; // 0..127 (H/16)
  const int mrow = lane & 15;
  int mi = m0 + mrow; if (mi > count - 1) mi = count - 1;  // clamp partial tile
  const int tok = ltok[e * Ntok + mi];
  const __bf16* aRow = x_bf + (size_t)tok * Ddim;
  const int kb = (lane < 16) ? 0 : 8;
  const int KC = Ddim / 32;                // 32 k-chunks
  const size_t bBase = (((size_t)e * (Hdim / 16) + ntile) * KC) * 512 + (size_t)lane * 16;
  v8f c1 = {}; v8f c2 = {};
  for (int kc = 0; kc < KC; ++kc) {
    const __bf16* ap = aRow + kc * 32 + kb;
    v8bf alo = *(const v8bf*)(ap);
    v8bf ahi = *(const v8bf*)(ap + 16);
    v16bf a;
#pragma unroll
    for (int i = 0; i < 8; ++i) { a[i] = alo[i]; a[i + 8] = ahi[i]; }
    const __bf16* bp1 = w1p + bBase + (size_t)kc * 512;
    const __bf16* bp2 = w2p + bBase + (size_t)kc * 512;
    v16bf bm1 = *(const v16bf*)bp1;
    v16bf bm2 = *(const v16bf*)bp2;
    if (kc + 1 < KC) {                      // pull next weight tiles toward L0
      __builtin_prefetch(bp1 + 512, 0, 1);
      __builtin_prefetch(bp2 + 512, 0, 1);
    }
    c1 = __builtin_amdgcn_wmma_f32_16x16x32_bf16(false, a, false, bm1, (short)0, c1, false, false);
    c2 = __builtin_amdgcn_wmma_f32_16x16x32_bf16(false, a, false, bm2, (short)0, c2, false, false);
  }
  // Epilogue: bias + SiLU * gate, store h (bf16). C layout: lane holds column
  // n = lane&15, rows v + (lane<16 ? 0 : 8) in VGPR v.
  const int n = ntile * 16 + (lane & 15);
  const float bv1 = b1[e * Hdim + n];
  const float bv2 = b2[e * Hdim + n];
  const int mhi = (lane < 16) ? 0 : 8;
#pragma unroll
  for (int v = 0; v < 8; ++v) {
    int m = v + mhi;
    if (m0 + m >= count) continue;
    int slot = lslot[e * Ntok + m0 + m];
    float s = c1[v] + bv1;
    float sig = 1.0f / (1.0f + __expf(-s));
    float hv = (s * sig) * (c2[v] + bv2);
    h_bf[(size_t)slot * Hdim + n] = f32_to_bf16_rne(hv);
  }
}

// ---------------------------------------------------------------------------
// 6) GEMM2: out[t] += wgt * (h @ w3[e] + b3[e]). Grid: (Ntok/16, D/128, E).
// Atomic f32 adds resolve the top-2 expert overlap per token.
// ---------------------------------------------------------------------------
__global__ void __launch_bounds__(256) moe_gemm2_kernel(
    const __bf16* __restrict__ h_bf, const __bf16* __restrict__ w3p,
    const float* __restrict__ b3, const int* __restrict__ cnt,
    const int* __restrict__ ltok, const int* __restrict__ lslot,
    const float* __restrict__ lw, float* __restrict__ out) {
  const int e = blockIdx.z;
  const int count = cnt[e];
  const int m0 = blockIdx.x * 16;
  if (m0 >= count) return;
  const int lane = threadIdx.x & 31;
  const int wave = threadIdx.x >> 5;
  const int ntile = blockIdx.y * 8 + wave; // 0..63 (D/16)
  const int mrow = lane & 15;
  int mi = m0 + mrow; if (mi > count - 1) mi = count - 1;
  const int slot_m = lslot[e * Ntok + mi];
  const __bf16* aRow = h_bf + (size_t)slot_m * Hdim;
  const int kb = (lane < 16) ? 0 : 8;
  const int KC = Hdim / 32;                // 64 k-chunks
  const size_t bBase = (((size_t)e * (Ddim / 16) + ntile) * KC) * 512 + (size_t)lane * 16;
  v8f c = {};
  for (int kc = 0; kc < KC; ++kc) {
    const __bf16* ap = aRow + kc * 32 + kb;
    v8bf alo = *(const v8bf*)(ap);
    v8bf ahi = *(const v8bf*)(ap + 16);
    v16bf a;
#pragma unroll
    for (int i = 0; i < 8; ++i) { a[i] = alo[i]; a[i + 8] = ahi[i]; }
    const __bf16* bp = w3p + bBase + (size_t)kc * 512;
    v16bf bm = *(const v16bf*)bp;
    if (kc + 1 < KC) __builtin_prefetch(bp + 512, 0, 1);
    c = __builtin_amdgcn_wmma_f32_16x16x32_bf16(false, a, false, bm, (short)0, c, false, false);
  }
  const int n = ntile * 16 + (lane & 15);
  const float bv = b3[e * Ddim + n];
  const int mhi = (lane < 16) ? 0 : 8;
#pragma unroll
  for (int v = 0; v < 8; ++v) {
    int m = v + mhi;
    if (m0 + m >= count) continue;
    int tok = ltok[e * Ntok + m0 + m];
    float wgt = lw[e * Ntok + m0 + m];
    atomicAdd(&out[(size_t)tok * Ddim + n], (c[v] + bv) * wgt);
  }
}

// ---------------------------------------------------------------------------
// Launch. Inputs (setup_inputs order): x, w1, b1, w2, b2, w3, b3, gate_w.
// Workspace layout (~143 MB): counters/lists, x_bf16, h_bf16, packed weights.
// ---------------------------------------------------------------------------
extern "C" void kernel_launch(void* const* d_in, const int* in_sizes, int n_in,
                              void* d_out, int out_size, void* d_ws, size_t ws_size,
                              hipStream_t stream) {
  const float* x  = (const float*)d_in[0];
  const float* w1 = (const float*)d_in[1];
  const float* b1 = (const float*)d_in[2];
  const float* w2 = (const float*)d_in[3];
  const float* b2 = (const float*)d_in[4];
  const float* w3 = (const float*)d_in[5];
  const float* b3 = (const float*)d_in[6];
  const float* gw = (const float*)d_in[7];
  float* out = (float*)d_out;

  char* ws = (char*)d_ws;
  size_t off = 0;
  int*    cnt   = (int*)(ws + off);    off += 256;
  int*    ltok  = (int*)(ws + off);    off += (size_t)Edim * Ntok * 4;
  int*    lslot = (int*)(ws + off);    off += (size_t)Edim * Ntok * 4;
  float*  lw    = (float*)(ws + off);  off += (size_t)Edim * Ntok * 4;
  __bf16* x_bf  = (__bf16*)(ws + off); off += (size_t)Ntok * Ddim * 2;
  __bf16* h_bf  = (__bf16*)(ws + off); off += (size_t)Ntok * 2 * Hdim * 2;
  __bf16* w1p   = (__bf16*)(ws + off); off += (size_t)Edim * Ddim * Hdim * 2;
  __bf16* w2p   = (__bf16*)(ws + off); off += (size_t)Edim * Ddim * Hdim * 2;
  __bf16* w3p   = (__bf16*)(ws + off); off += (size_t)Edim * Hdim * Ddim * 2;
  (void)ws_size; (void)in_sizes; (void)n_in; (void)out_size;

  const size_t nOut = (size_t)Ntok * Ddim;          // 4.19M f32
  const size_t nW   = (size_t)Edim * Ddim * Hdim;   // 16.8M per weight

  moe_init_kernel<<<(unsigned)((nOut + 255) / 256), 256, 0, stream>>>(cnt, out, nOut);
  moe_cvt_x_kernel<<<(unsigned)((nOut + 255) / 256), 256, 0, stream>>>(x, x_bf, nOut);
  moe_pack_w_kernel<<<(unsigned)((nW + 255) / 256), 256, 0, stream>>>(w1, w1p, Ddim, Hdim);
  moe_pack_w_kernel<<<(unsigned)((nW + 255) / 256), 256, 0, stream>>>(w2, w2p, Ddim, Hdim);
  moe_pack_w_kernel<<<(unsigned)((nW + 255) / 256), 256, 0, stream>>>(w3, w3p, Hdim, Ddim);
  moe_gate_kernel<<<Ntok / 256, 256, 0, stream>>>(x, gw, cnt, ltok, lslot, lw);
  moe_gemm1_kernel<<<dim3(Ntok / 16, Hdim / 128, Edim), 256, 0, stream>>>(
      x_bf, w1p, w2p, b1, b2, cnt, ltok, lslot, h_bf);
  moe_gemm2_kernel<<<dim3(Ntok / 16, Ddim / 128, Edim), 256, 0, stream>>>(
      h_bf, w3p, b3, cnt, ltok, lslot, lw, out);
}
